// Network_39977555591763
// MI455X (gfx1250) — matrix operand order repaired and checked by reading, not compiled, not run on hardware
//
#include <hip/hip_runtime.h>

// ---- CDNA5 (gfx1250) wave32 WMMA types ----
typedef _Float16 half4 __attribute__((ext_vector_type(4)));
typedef _Float16 half8 __attribute__((ext_vector_type(8)));
typedef _Float16 v16h  __attribute__((ext_vector_type(16)));
typedef float    v8f   __attribute__((ext_vector_type(8)));

#define B_    100
#define T_    500
#define CIN   300
#define COUT  500
#define C4    75            // CIN / 4 (float4 granules)
#define KPAD  328           // LDS row pitch in f16: 656B = 164 DWORDs -> conflict-free b128
#define KTAIL (KPAD - CIN)  // 28 zero-pad halfs per row
#define OW    64            // o-columns per workgroup (4 waves x 16)
#define TT    16            // t-chunk (WMMA M)
#define NCHUNK 32           // ceil(500/16)
#define NKS   10            // K steps of 32 (K padded 300 -> 320)

// A (16x32 f16) fragment load from LDS: lane row M = laneLo;
// K = ks*32 + {laneHi*8 .. +7} and the same +16 (ISA 7.12.2).
__device__ __forceinline__ v16h load_a_frag(const _Float16* sX, int laneLo, int laneHi, int ks) {
    const _Float16* p = &sX[laneLo * KPAD + ks * 32 + laneHi * 8];
    half8 lo = *(const half8*)(p);
    half8 hi = *(const half8*)(p + 16);
    return __builtin_shufflevector(lo, hi,
        0,1,2,3,4,5,6,7,8,9,10,11,12,13,14,15);
}

__global__ __launch_bounds__(128) void lif_fused_kernel(
    const float* __restrict__ x,      // [B,T,CIN]
    const float* __restrict__ W,      // [COUT,CIN]
    const float* __restrict__ alpha,  // [B*COUT]
    float* __restrict__ out)          // [B,T,COUT]
{
    __shared__ __align__(16) float    sScan[TT * OW];      //  4,096 B
    __shared__ __align__(16) _Float16 sX[TT * KPAD];       // 10,496 B
    __shared__ __align__(16) _Float16 sW[OW * KPAD];       // 41,984 B

    const int tid    = threadIdx.x;
    const int wv     = tid >> 5;        // wave 0..3 -> o sub-tile
    const int lane   = tid & 31;
    const int laneLo = lane & 15;
    const int laneHi = lane >> 4;       // 0 or 1
    const int b      = blockIdx.y;
    const int o0     = blockIdx.x * OW;

    // ---- zero-fill the static K-pad columns (c in [300,328)) once ----
    for (int i = tid; i < TT * KTAIL; i += 128) {
        const int r = i / KTAIL, c = CIN + (i - r * KTAIL);
        sX[r * KPAD + c] = (_Float16)0.0f;
    }
    for (int i = tid; i < OW * KTAIL; i += 128) {
        const int r = i / KTAIL, c = CIN + (i - r * KTAIL);
        sW[r * KPAD + c] = (_Float16)0.0f;
    }

    // ---- stage W slice (f32 -> f16) into LDS, vectorized float4 loads ----
    for (int i = tid; i < OW * C4; i += 128) {
        const int ol = i / C4;
        const int c4 = i - ol * C4;
        const int o  = o0 + ol;
        float4 v = make_float4(0.f, 0.f, 0.f, 0.f);
        if (o < COUT) v = *(const float4*)&W[o * CIN + c4 * 4];
        half4 h = { (_Float16)v.x, (_Float16)v.y, (_Float16)v.z, (_Float16)v.w };
        *(half4*)&sW[ol * KPAD + c4 * 4] = h;
    }
    __syncthreads();

    // ---- hoist this wave's 10 B fragments into registers (reused all chunks) ----
    // B (32x16 f16): lane holds column N = laneLo of o sub-tile;
    // K = ks*32 + {laneHi*8 .. +7} and the same +16 (matches A fragment order).
    v16h bfrag[NKS];
    {
        const int orow = wv * 16 + laneLo;
        #pragma unroll
        for (int ks = 0; ks < NKS; ++ks) {
            const _Float16* p = &sW[orow * KPAD + ks * 32 + laneHi * 8];
            half8 lo = *(const half8*)(p);
            half8 hi = *(const half8*)(p + 16);
            bfrag[ks] = __builtin_shufflevector(lo, hi,
                0,1,2,3,4,5,6,7,8,9,10,11,12,13,14,15);
        }
    }

    // ---- per-neuron LIF state for the 64 scan threads ----
    const int o_my  = o0 + (tid & (OW - 1));
    float a_dec = 0.0f, v_mem = 0.0f;
    if (tid < OW && o_my < COUT) a_dec = alpha[b * COUT + o_my];

    for (int chunk = 0; chunk < NCHUNK; ++chunk) {
        const int t0 = chunk * TT;
        __syncthreads();   // sScan consumed / sX reusable

        // stage x tile [16 t][CIN] f32 -> f16, vectorized float4 -> half4
        for (int i = tid; i < TT * C4; i += 128) {
            const int r  = i / C4;
            const int c4 = i - r * C4;
            const int t  = t0 + r;
            float4 v = make_float4(0.f, 0.f, 0.f, 0.f);
            if (t < T_) v = *(const float4*)&x[(b * T_ + t) * CIN + c4 * 4];
            half4 h = { (_Float16)v.x, (_Float16)v.y, (_Float16)v.z, (_Float16)v.w };
            *(half4*)&sX[r * KPAD + c4 * 4] = h;
        }
        __syncthreads();

        // prefetch next chunk's x rows into cache (global_prefetch_b8)
        if (tid < TT) {
            const int tn = t0 + TT + tid;
            if (tn < T_) __builtin_prefetch(&x[(b * T_ + tn) * CIN], 0, 3);
        }

        // ---- K loop, software-pipelined depth 1: loads for ks+1 issue before
        //      the WMMA of ks, so only 2 ds_loads need to be waited at a time ----
        v8f acc = {};
        v16h a_cur = load_a_frag(sX, laneLo, laneHi, 0);
        #pragma unroll
        for (int ks = 0; ks < NKS; ++ks) {
            v16h a_nxt = a_cur;
            if (ks + 1 < NKS) a_nxt = load_a_frag(sX, laneLo, laneHi, ks + 1);
            acc = __builtin_amdgcn_wmma_f32_16x16x32_f16(
                false, a_cur, false, bfrag[ks], (short)0, acc, false, false);
            a_cur = a_nxt;
        }

        // ---- spill C tile to LDS: VGPR r -> M = r + 8*laneHi, N = laneLo ----
        {
            const int col = wv * 16 + laneLo;
            #pragma unroll
            for (int r = 0; r < 8; ++r)
                sScan[(r + 8 * laneHi) * OW + col] = acc[r];
        }
        __syncthreads();

        // ---- LIF scan: 64 threads, one o-column each, 16 sequential timesteps ----
        if (tid < OW) {
            float xv[TT];
            #pragma unroll
            for (int m = 0; m < TT; ++m) xv[m] = sScan[m * OW + tid];
            #pragma unroll
            for (int m = 0; m < TT; ++m) {
                float vp = fmaxf(a_dec * v_mem + xv[m], -1.0f);   // leak+integrate, min_v_mem=-1
                float s  = (vp >= 1.0f) ? floorf(vp) : 0.0f;       // multi-spike, threshold=1
                v_mem = vp - s;                                    // reset by subtraction
                const int t = t0 + m;
                if (t < T_ && o_my < COUT)
                    out[(b * T_ + t) * COUT + o_my] = s;
            }
        }
    }
}

extern "C" void kernel_launch(void* const* d_in, const int* in_sizes, int n_in,
                              void* d_out, int out_size, void* d_ws, size_t ws_size,
                              hipStream_t stream) {
    (void)in_sizes; (void)n_in; (void)d_ws; (void)ws_size; (void)out_size;
    const float* x     = (const float*)d_in[0];   // [100,500,300]
    const float* W     = (const float*)d_in[1];   // [500,300]
    const float* alpha = (const float*)d_in[2];   // [50000]
    float* out         = (float*)d_out;           // [100,500,500]

    dim3 grid((COUT + OW - 1) / OW, B_);          // (8, 100)
    dim3 block(128);                              // 4 wave32
    lif_fused_kernel<<<grid, block, 0, stream>>>(x, W, alpha, out);
}
